// NT_Xent_loss_56461640073786
// MI455X (gfx1250) — matrix operand order repaired and checked by reading, not compiled, not run on hardware
//
#include <hip/hip_runtime.h>

#define N_HALF 4096
#define TWO_N  8192
#define DIM    128
#define TAU_INV 2.0f            // 1/0.5
#define LOG2E  1.4426950408889634f

typedef __attribute__((ext_vector_type(16))) _Float16 v16h;
typedef __attribute__((ext_vector_type(8)))  _Float16 v8h;
typedef __attribute__((ext_vector_type(4)))  _Float16 v4h;
typedef __attribute__((ext_vector_type(8)))  float    v8f;
typedef __attribute__((ext_vector_type(4)))  unsigned int v4u;
typedef __attribute__((ext_vector_type(8)))  int      v8i;
typedef __attribute__((ext_vector_type(4)))  int      v4i;

#if __has_builtin(__builtin_amdgcn_tensor_load_to_lds) && __has_builtin(__builtin_amdgcn_s_wait_tensorcnt)
#define USE_TDM 1
#else
#define USE_TDM 0
#endif

__device__ inline v8f wmma16(v16h a, v16h b, v8f c) {
    // D = A(16x32 f16) * B(32x16 f16) + C(16x16 f32)
    return __builtin_amdgcn_wmma_f32_16x16x32_f16(
        false, a, false, b, (short)0, c, false, false);
}

// ---------------------------------------------------------------------------
// Kernel 1: cosine-normalize rows of concat(z_i, z_j); emit f16 hi/lo split.
// ---------------------------------------------------------------------------
__global__ __launch_bounds__(256) void nt_xent_normalize(
    const float* __restrict__ zi, const float* __restrict__ zj,
    _Float16* __restrict__ phi, _Float16* __restrict__ plo)
{
    const int wave = threadIdx.x >> 5;
    const int lane = threadIdx.x & 31;
    const int row  = blockIdx.x * 8 + wave;          // 1024 blocks -> 8192 rows

    const float* src = (row < N_HALF) ? (zi + (size_t)row * DIM)
                                      : (zj + (size_t)(row - N_HALF) * DIM);
    float4 v = *(const float4*)(src + lane * 4);

    float ssq = v.x * v.x + v.y * v.y + v.z * v.z + v.w * v.w;
    #pragma unroll
    for (int m = 16; m >= 1; m >>= 1) ssq += __shfl_xor(ssq, m);

    const float scale = 1.0f / fmaxf(sqrtf(ssq), 1e-8f);

    float xs[4] = { v.x * scale, v.y * scale, v.z * scale, v.w * scale };
    v4h hi4, lo4;
    #pragma unroll
    for (int j = 0; j < 4; ++j) {
        float    x = xs[j];
        _Float16 h = (_Float16)x;
        hi4[j] = h;
        lo4[j] = (_Float16)(x - (float)h);           // residual, ~2^-11 * x
    }
    *(v4h*)(phi + (size_t)row * DIM + lane * 4) = hi4;
    *(v4h*)(plo + (size_t)row * DIM + lane * 4) = lo4;
}

// ---------------------------------------------------------------------------
// Kernel 2: fused  sim = (pn pn^T)/tau  +  row-wise shifted logsumexp + pos.
// 128 WGs x 256 thr; WG owns 64 rows, sweeps 8192 cols in 64-wide tiles.
// B tiles are DMA'd by the Tensor Data Mover into double-buffered LDS
// (TDM pad feature reproduces the +16B/row bank-spread padding), overlapped
// with WMMA compute on the other buffer.  A fragments register-cached.
// ---------------------------------------------------------------------------
#define MT      64            // rows per workgroup
#define NT      64            // columns per tile
#define NTILES  (TWO_N / NT)  // 128
#define LDSTR   136           // half-element row stride (128 + 8 pad = 272 B)

#if USE_TDM
// One 2D-tile TDM descriptor: NT rows x DIM halfs from row-major f16 matrix,
// padded in LDS by 16 B after every 256 B row.
__device__ inline void tdm_load_tile(const _Float16* src, void* lds_dst) {
    const unsigned lds_addr = (unsigned)(size_t)lds_dst;     // low 32 = LDS offset
    const unsigned long long ga = (unsigned long long)(size_t)src;

    v4u g0;
    g0[0] = 1u;                                   // count=1, is_restore=0, no gather
    g0[1] = lds_addr;                             // D#.lds_addr  (bits 63:32)
    g0[2] = (unsigned)ga;                         // global_addr[31:0]
    g0[3] = (unsigned)((ga >> 32) & 0x1FFFFFFull) // global_addr[56:32]
          | (2u << 30);                           // type = 2 ("image")

    v8i g1;
    g1[0] = (1 << 16)                             // data_size = 2 bytes
          | (1 << 20)                             // pad_enable
          | (5 << 22)                             // pad_interval: 64 dwords (=256B row)
          | (3 << 25);                            // pad_amount:   4 dwords (=16B)
    g1[1] = (DIM   & 0xFFFF) << 16;               // tensor_dim0[15:0]  (bits 63:48)
    g1[2] = (TWO_N & 0xFFFF) << 16;               // tensor_dim1[15:0]  (bits 95:80)
    g1[3] = (DIM   & 0xFFFF) << 16;               // tile_dim0 = 128    (bits 127:112)
    g1[4] = NT;                                   // tile_dim1 = 64, tile_dim2 = 0
    g1[5] = DIM;                                  // tensor_dim0_stride = 128
    g1[6] = 0;                                    // dim0_stride hi / dim1_stride lo
    g1[7] = 0;                                    // dim1_stride hi
    v4i z4 = {0, 0, 0, 0};
#if defined(__clang_major__) && __clang_major__ >= 23
    v8i z8 = {0, 0, 0, 0, 0, 0, 0, 0};
    __builtin_amdgcn_tensor_load_to_lds(g0, g1, z4, z4, z8, 0);
#else
    __builtin_amdgcn_tensor_load_to_lds(g0, g1, z4, z4, 0);
#endif
}
#endif

__global__ __launch_bounds__(256) void nt_xent_main(
    const _Float16* __restrict__ phi, const _Float16* __restrict__ plo,
    float* __restrict__ row_loss)
{
#if USE_TDM
    __shared__ _Float16 Bhi[2][NT][LDSTR];
    __shared__ _Float16 Blo[2][NT][LDSTR];
#else
    __shared__ _Float16 Bhi[1][NT][LDSTR];
    __shared__ _Float16 Blo[1][NT][LDSTR];
#endif
    __shared__ float    sumbuf[2][MT];
    __shared__ float    posbuf[2][MT];

    const int tid  = threadIdx.x;
    const int w    = tid >> 5;
    const int lane = tid & 31;
    const int mf   = w & 3;          // row fragment 0..3
    const int h    = w >> 2;         // column half 0..1
    const int g    = lane >> 4;      // lane half within wave
    const int ml   = lane & 15;

    const int rowbase = blockIdx.x * MT;

    // ---- register-cache A fragments (hi & lo) for this wave's 16 rows.
    // A layout (16-bit 16x32): lane = row, elems 0..15 = K kc*32+g*16..
    // (16 contiguous K, matching B)  -> two 16 B global loads per fragment.
    const int arow = rowbase + mf * 16 + ml;
    v16h afh[4], afl[4];
    #pragma unroll
    for (int kc = 0; kc < 4; ++kc) {
        const int k0 = kc * 32 + g * 16;
        v8h h0 = *(const v8h*)(phi + (size_t)arow * DIM + k0);
        v8h h1 = *(const v8h*)(phi + (size_t)arow * DIM + k0 + 8);
        v8h l0 = *(const v8h*)(plo + (size_t)arow * DIM + k0);
        v8h l1 = *(const v8h*)(plo + (size_t)arow * DIM + k0 + 8);
        afh[kc] = __builtin_shufflevector(h0, h1, 0,1,2,3,4,5,6,7,8,9,10,11,12,13,14,15);
        afl[kc] = __builtin_shufflevector(l0, l1, 0,1,2,3,4,5,6,7,8,9,10,11,12,13,14,15);
    }

    v8f sumexp = {0.f,0.f,0.f,0.f,0.f,0.f,0.f,0.f};
    v8f posacc = {0.f,0.f,0.f,0.f,0.f,0.f,0.f,0.f};

#if USE_TDM
    if (w == 0) {                                   // prologue: tile 0 -> buffer 0
        tdm_load_tile(phi, &Bhi[0][0][0]);
        tdm_load_tile(plo, &Blo[0][0][0]);
    }
#else
    const int bc = tid >> 2;            // column within tile
    const int bk = (tid & 3) * 32;      // K quarter
#endif

    for (int tile = 0; tile < NTILES; ++tile) {
#if USE_TDM
        const int buf = tile & 1;
        if (w == 0) {
            if (tile + 1 < NTILES) {
                // Overwriting buf^1 is safe: reaching here means every wave
                // passed the end-of-compute barrier of tile-1 (which read buf^1).
                const _Float16* nsrc = phi + (size_t)(tile + 1) * NT * DIM;
                tdm_load_tile(nsrc, &Bhi[buf ^ 1][0][0]);
                tdm_load_tile(plo + (size_t)(tile + 1) * NT * DIM, &Blo[buf ^ 1][0][0]);
                __builtin_amdgcn_s_wait_tensorcnt(2);   // current tile's 2 DMAs done
            } else {
                __builtin_amdgcn_s_wait_tensorcnt(0);
            }
        }
        __syncthreads();                               // tile data visible to all
#else
        const int buf = 0;
        __syncthreads();                               // prev tile fully consumed
        {
            const size_t srcOff = ((size_t)(tile * NT + bc)) * DIM + bk;
            #pragma unroll
            for (int i = 0; i < 4; ++i) {
                *(v8h*)&Bhi[0][bc][bk + i * 8] = *(const v8h*)(phi + srcOff + i * 8);
                *(v8h*)&Blo[0][bc][bk + i * 8] = *(const v8h*)(plo + srcOff + i * 8);
            }
            if (tile + 1 < NTILES) {
                const size_t nOff = ((size_t)((tile + 1) * NT + bc)) * DIM + bk;
                __builtin_prefetch(phi + nOff, 0, 0);
                __builtin_prefetch(plo + nOff, 0, 0);
            }
        }
        __syncthreads();                               // B tile ready
#endif

        #pragma unroll
        for (int q = 0; q < 2; ++q) {
            const int nf  = h * 2 + q;                 // 16-col fragment in tile
            const int col = nf * 16 + ml;              // B: lane = column
            v8f acc = {0.f,0.f,0.f,0.f,0.f,0.f,0.f,0.f};
            #pragma unroll
            for (int kc = 0; kc < 4; ++kc) {
                const int kb = kc * 32 + g * 16;       // 16 contiguous K per lane
                v16h bh = *(const v16h*)&Bhi[buf][col][kb];
                v16h bl = *(const v16h*)&Blo[buf][col][kb];
                acc = wmma16(afh[kc], bh, acc);        // hi*hi
                acc = wmma16(afh[kc], bl, acc);        // hi*lo
                acc = wmma16(afl[kc], bh, acc);        // lo*hi
            }
            // epilogue: fixed-shift softmax terms + positive-pair capture
            const int C = tile * NT + col;             // global column
            #pragma unroll
            for (int v = 0; v < 8; ++v) {
                const int R   = rowbase + mf * 16 + v + 8 * g;  // C/D: M = v + 8*g
                const float d = acc[v];
                // exp(sim/tau - 4) == exp2(d * 2*log2e - 4*log2e); sim/tau in [-4,4]
                float e = exp2f(fmaf(d, TAU_INV * LOG2E, -4.0f * LOG2E));
                if (C == R) e = 0.0f;                  // mask diagonal
                sumexp[v] += e;
                if (C == (R ^ N_HALF)) posacc[v] += d * TAU_INV;
            }
        }
        __syncthreads();                               // compute done; buf reusable
    }

    // reduce across the 16 lanes of each half-wave (rows live per half)
    #pragma unroll
    for (int m = 1; m < 16; m <<= 1) {
        #pragma unroll
        for (int v = 0; v < 8; ++v) {
            sumexp[v] += __shfl_xor(sumexp[v], m);
            posacc[v] += __shfl_xor(posacc[v], m);
        }
    }
    if (ml == 0) {
        #pragma unroll
        for (int v = 0; v < 8; ++v) {
            sumbuf[h][mf * 16 + v + 8 * g] = sumexp[v];
            posbuf[h][mf * 16 + v + 8 * g] = posacc[v];
        }
    }
    __syncthreads();

    if (tid < MT) {
        const float tot = sumbuf[0][tid] + sumbuf[1][tid];
        const float pos = posbuf[0][tid] + posbuf[1][tid];
        row_loss[rowbase + tid] = logf(tot) + 4.0f - pos;   // lse - sim[r,partner]
    }
}

// ---------------------------------------------------------------------------
// Kernel 3: deterministic fixed-order reduction of 8192 row losses.
// ---------------------------------------------------------------------------
__global__ __launch_bounds__(256) void nt_xent_reduce(
    const float* __restrict__ row_loss, float* __restrict__ out)
{
    __shared__ float sb[256];
    float s = 0.0f;
    for (int i = threadIdx.x; i < TWO_N; i += 256) s += row_loss[i];
    sb[threadIdx.x] = s;
    __syncthreads();
    for (int k = 128; k > 0; k >>= 1) {
        if (threadIdx.x < k) sb[threadIdx.x] += sb[threadIdx.x + k];
        __syncthreads();
    }
    if (threadIdx.x == 0) out[0] = sb[0] / (float)TWO_N;
}

// ---------------------------------------------------------------------------
extern "C" void kernel_launch(void* const* d_in, const int* in_sizes, int n_in,
                              void* d_out, int out_size, void* d_ws, size_t ws_size,
                              hipStream_t stream)
{
    (void)in_sizes; (void)n_in; (void)out_size; (void)ws_size;
    const float* zi = (const float*)d_in[0];
    const float* zj = (const float*)d_in[1];

    _Float16* phi = (_Float16*)d_ws;
    _Float16* plo = phi + (size_t)TWO_N * DIM;
    float* row_loss = (float*)((char*)d_ws + 2 * (size_t)TWO_N * DIM * sizeof(_Float16));
    float* out = (float*)d_out;

    nt_xent_normalize<<<TWO_N / 8, 256, 0, stream>>>(zi, zj, phi, plo);
    nt_xent_main<<<TWO_N / MT, 256, 0, stream>>>(phi, plo, row_loss);
    nt_xent_reduce<<<1, 256, 0, stream>>>(row_loss, out);
}
